// CATDecoderAttention_38319698215053
// MI455X (gfx1250) — matrix operand order repaired and checked by reading, not compiled, not run on hardware
//
#include <hip/hip_runtime.h>
#include <hip/hip_bf16.h>

#define B_   2
#define T_   2048
#define HID_ 1024
#define H_   16
#define HK_  4
#define D_   64
#define G_   (H_ / HK_)

typedef __attribute__((ext_vector_type(16))) __bf16 bf16x16;
typedef __attribute__((ext_vector_type(8)))  float  floatx8;

union FragU { bf16x16 v; unsigned short u[16]; };

__device__ __forceinline__ unsigned short f2bf(float x) {
  union { float f; unsigned int i; } c; c.f = x;
  unsigned int u = c.i;
  u += 0x7FFFu + ((u >> 16) & 1u);           // round-to-nearest-even
  return (unsigned short)(u >> 16);
}

// Generic shared-memory pointer -> LDS byte offset (LDS aperture keeps the
// offset in addr[31:0]).
__device__ __forceinline__ unsigned lds_off_of(const void* p) {
  return (unsigned)(unsigned long long)p;
}

// CDNA5 async global->LDS copy, 16 bytes per lane, tracked with ASYNCcnt.
__device__ __forceinline__ void async_copy_b128(unsigned lds_byte_off, const void* gaddr) {
  asm volatile("global_load_async_to_lds_b128 %0, %1, off"
               :: "v"(lds_byte_off), "v"(gaddr)
               : "memory");
}

__device__ __forceinline__ void wait_async0() {
#if __has_builtin(__builtin_amdgcn_s_wait_asynccnt)
  __builtin_amdgcn_s_wait_asynccnt(0);
#else
  asm volatile("s_wait_asynccnt 0x0" ::: "memory");
#endif
}

// A-fragment (16x32 bf16, MxK): lane m=l%16; h=l/16; elem i -> K=h*16+i.
// One contiguous 16-element LDS read.
__device__ __forceinline__ bf16x16 frag_a(const unsigned short* p, int row0, int k0, int ld) {
  int lane = threadIdx.x & 31;
  int m = lane & 15, h = lane >> 4;
  const unsigned short* base = p + (size_t)(row0 + m) * ld + k0;
  FragU f;
#pragma unroll
  for (int i = 0; i < 16; ++i) f.u[i] = base[h * 16 + i];
  return f.v;
}

// B-fragment (32x16 bf16, KxN) gathered from a row-major [N][K] tile:
// lane n=l%16; h=l/16; elem i -> K = h*16+i. One contiguous 16-element read.
__device__ __forceinline__ bf16x16 frag_b(const unsigned short* p, int n0, int k0, int ld) {
  int lane = threadIdx.x & 31;
  int n = lane & 15, h = lane >> 4;
  const unsigned short* base = p + (size_t)(n0 + n) * ld + k0 + h * 16;
  FragU f;
#pragma unroll
  for (int i = 0; i < 16; ++i) f.u[i] = base[i];
  return f.v;
}

// B-fragment from a row-major [K][N] tile (used for V in P*V): strided reads.
__device__ __forceinline__ bf16x16 frag_b_kn(const unsigned short* p, int n0, int ld) {
  int lane = threadIdx.x & 31;
  int n = lane & 15, h = lane >> 4;
  FragU f;
#pragma unroll
  for (int i = 0; i < 16; ++i) f.u[i] = p[(size_t)(h * 16 + i) * ld + n0 + n];
  return f.v;
}

__device__ __forceinline__ floatx8 wmma_bf16(bf16x16 a, bf16x16 b, floatx8 c) {
  return __builtin_amdgcn_wmma_f32_16x16x32_bf16(false, a, false, b, (short)0, c, false, false);
}

// ---------------------------------------------------------------------------
// f32 -> bf16 bulk convert, 4 elements per thread (one b64 store).
// ---------------------------------------------------------------------------
__global__ __launch_bounds__(256) void cvt_f32_bf16(const float* __restrict__ in,
                                                    unsigned short* __restrict__ out,
                                                    int n4) {
  int i = blockIdx.x * blockDim.x + threadIdx.x;
  if (i >= n4) return;
  float4 f = ((const float4*)in)[i];
  union { unsigned short u[4]; unsigned long long q; } r;
  r.u[0] = f2bf(f.x); r.u[1] = f2bf(f.y); r.u[2] = f2bf(f.z); r.u[3] = f2bf(f.w);
  ((unsigned long long*)out)[i] = r.q;
}

// ---------------------------------------------------------------------------
// C[M,N] = A[M,K] (bf16) * W[N,K]^T (bf16), f32 out. Block tile 128x64, BK=32,
// 256 threads = 8 waves (4x2), each wave a 32x32 sub-tile = 2x2 WMMAs.
// Tiles staged with async global->LDS B128 copies (ASYNCcnt).
// ---------------------------------------------------------------------------
#define BM 128
#define BN 64
#define BK 32

__global__ __launch_bounds__(256) void gemm_bf16_wmma(const unsigned short* __restrict__ A,
                                                      const unsigned short* __restrict__ W,
                                                      float* __restrict__ C,
                                                      int M, int N, int K) {
  __shared__ unsigned short As[BM * BK];   // 8 KB
  __shared__ unsigned short Bs[BN * BK];   // 4 KB
  const int m0 = blockIdx.x * BM;
  const int n0 = blockIdx.y * BN;
  const int tid  = threadIdx.x;
  const int wave = tid >> 5;
  const int wm = wave >> 1, wn = wave & 1;
  const unsigned asb = lds_off_of(As);
  const unsigned bsb = lds_off_of(Bs);

  floatx8 acc[2][2] = {};

  for (int k0 = 0; k0 < K; k0 += BK) {
    // A tile: 128x32 bf16 = 8 KB = 512 x b128 -> 2 per thread
#pragma unroll
    for (int i = 0; i < 2; ++i) {
      int idx = tid + i * 256;
      int r = idx >> 2, c8 = (idx & 3) * 8;           // BK/8 = 4 chunks per row
      async_copy_b128(asb + idx * 16, A + (size_t)(m0 + r) * K + k0 + c8);
    }
    // B tile: 64x32 bf16 = 4 KB = 256 x b128 -> 1 per thread
    {
      int r = tid >> 2, c8 = (tid & 3) * 8;
      async_copy_b128(bsb + tid * 16, W + (size_t)(n0 + r) * K + k0 + c8);
    }
    if (k0 + BK < K) {                                // warm L2 for next k-tile
      __builtin_prefetch(A + (size_t)(m0 + (tid >> 2)) * K + k0 + BK, 0, 1);
      __builtin_prefetch(W + (size_t)(n0 + (tid & 63)) * K + k0 + BK, 0, 1);
    }
    wait_async0();
    __syncthreads();

    bf16x16 a0 = frag_a(As, wm * 32,      0, BK);
    bf16x16 a1 = frag_a(As, wm * 32 + 16, 0, BK);
    bf16x16 b0 = frag_b(Bs, wn * 32,      0, BK);
    bf16x16 b1 = frag_b(Bs, wn * 32 + 16, 0, BK);
    acc[0][0] = wmma_bf16(a0, b0, acc[0][0]);
    acc[0][1] = wmma_bf16(a0, b1, acc[0][1]);
    acc[1][0] = wmma_bf16(a1, b0, acc[1][0]);
    acc[1][1] = wmma_bf16(a1, b1, acc[1][1]);
    __syncthreads();
  }

  const int lane = tid & 31;
  const int nn = lane & 15, hh = lane >> 4;
  float* cp = C + (size_t)(m0 + wm * 32 + hh * 8) * N + n0 + wn * 32 + nn;
#pragma unroll
  for (int i = 0; i < 2; ++i)
#pragma unroll
    for (int j = 0; j < 2; ++j)
#pragma unroll
      for (int r = 0; r < 8; ++r)
        cp[(size_t)(i * 16 + r) * N + j * 16] = acc[i][j][r];
}

// ---------------------------------------------------------------------------
// RMSNorm + RoPE + bf16 relayout. One thread per (b,t,head) row of D=64.
// ---------------------------------------------------------------------------
__global__ __launch_bounds__(256) void norm_rope_relayout(
    const float* __restrict__ qf, const float* __restrict__ kf, const float* __restrict__ vf,
    const float* __restrict__ cosp, const float* __restrict__ sinp,
    const float* __restrict__ qw, const float* __restrict__ kw,
    unsigned short* __restrict__ Qb, unsigned short* __restrict__ Kb,
    unsigned short* __restrict__ Vb) {
  const int QROWS = B_ * T_ * H_;
  const int KROWS = B_ * T_ * HK_;
  int row = blockIdx.x * blockDim.x + threadIdx.x;

  const float* src; const float* w; unsigned short* dst;
  int t, doNorm;
  if (row < QROWS) {
    int h = row % H_; int tmp = row / H_; t = tmp % T_; int b = tmp / T_;
    src = qf + (size_t)(b * T_ + t) * HID_ + h * D_;
    dst = Qb + ((size_t)(b * H_ + h) * T_ + t) * D_;
    w = qw; doNorm = 1;
  } else if (row < QROWS + KROWS) {
    int idx = row - QROWS;
    int hk = idx % HK_; int tmp = idx / HK_; t = tmp % T_; int b = tmp / T_;
    src = kf + (size_t)(b * T_ + t) * (HK_ * D_) + hk * D_;
    dst = Kb + ((size_t)(b * HK_ + hk) * T_ + t) * D_;
    w = kw; doNorm = 1;
  } else if (row < QROWS + 2 * KROWS) {
    int idx = row - QROWS - KROWS;
    int hk = idx % HK_; int tmp = idx / HK_; t = tmp % T_; int b = tmp / T_;
    src = vf + (size_t)(b * T_ + t) * (HK_ * D_) + hk * D_;
    dst = Vb + ((size_t)(b * HK_ + hk) * T_ + t) * D_;
    w = nullptr; doNorm = 0;
  } else {
    return;
  }

  if (!doNorm) {
#pragma unroll
    for (int d = 0; d < D_; ++d) dst[d] = f2bf(src[d]);
    return;
  }

  float v[D_];
  float ss = 0.f;
#pragma unroll
  for (int d = 0; d < D_; ++d) { float x = src[d]; v[d] = x; ss += x * x; }
  float inv = rsqrtf(ss * (1.0f / D_) + 1e-6f);
#pragma unroll
  for (int j = 0; j < D_ / 2; ++j) {
    float x1 = v[j] * inv * w[j];
    float x2 = v[j + D_ / 2] * inv * w[j + D_ / 2];
    float c = cosp[(size_t)t * (D_ / 2) + j];
    float s = sinp[(size_t)t * (D_ / 2) + j];
    dst[j]          = f2bf(x1 * c - x2 * s);
    dst[j + D_ / 2] = f2bf(x1 * s + x2 * c);
  }
}

// ---------------------------------------------------------------------------
// Causal flash attention with GQA. Block = 128 queries of one (b,h);
// 8 waves x 16 query rows. Key tiles of 32. bf16 WMMA for S and PV.
// Q/K/V tiles staged with async global->LDS copies. O written bf16.
// ---------------------------------------------------------------------------
__global__ __launch_bounds__(256) void attn_wmma(const unsigned short* __restrict__ Qb,
                                                 const unsigned short* __restrict__ Kb,
                                                 const unsigned short* __restrict__ Vb,
                                                 unsigned short* __restrict__ O) {
  __shared__ unsigned short Qs[128 * 64];      // 16 KB [q][d]
  __shared__ unsigned short Ks[32 * 64];       //  4 KB [key][d]
  __shared__ unsigned short Vs[32 * 64];       //  4 KB [key][d]
  __shared__ unsigned short Ps[8][16 * 32];    //  8 KB per-wave P scratch

  const int bh = blockIdx.y;
  const int b = bh / H_, h = bh % H_, hk = h / G_;
  const int q0 = blockIdx.x * 128;
  const unsigned short* Qp = Qb + ((size_t)(b * H_ + h) * T_ + q0) * D_;
  const unsigned short* Kp = Kb + ((size_t)(b * HK_ + hk) * T_) * D_;
  const unsigned short* Vp = Vb + ((size_t)(b * HK_ + hk) * T_) * D_;

  const int tid = threadIdx.x, wave = tid >> 5, lane = tid & 31;
  const int nn = lane & 15, hh = lane >> 4;
  const unsigned qsb = lds_off_of(Qs);
  const unsigned ksb = lds_off_of(Ks);
  const unsigned vsb = lds_off_of(Vs);

  // Q tile: contiguous 16 KB -> 1024 x b128, 4 per thread (async).
#pragma unroll
  for (int i = 0; i < 4; ++i) {
    int idx = tid + i * 256;
    async_copy_b128(qsb + idx * 16, (const char*)Qp + (size_t)idx * 16);
  }

  float m[8], l[8];
  floatx8 o[4] = {};
#pragma unroll
  for (int r = 0; r < 8; ++r) { m[r] = -3.0e38f; l[r] = 0.f; }

  const int qend = q0 + 127;
  const float scale = 0.125f;                   // 1/sqrt(64)

  for (int k0 = 0; k0 <= qend; k0 += 32) {
    __syncthreads();                            // prior iter done reading Ks/Vs
    // K and V tiles: contiguous 4 KB each -> 256 x b128, 1 per thread each.
    async_copy_b128(ksb + tid * 16, (const char*)(Kp + (size_t)k0 * D_) + (size_t)tid * 16);
    async_copy_b128(vsb + tid * 16, (const char*)(Vp + (size_t)k0 * D_) + (size_t)tid * 16);
    wait_async0();
    __syncthreads();

    // S = Q * K^T  (16 rows x 32 keys per wave)
    floatx8 s[2] = {};
#pragma unroll
    for (int kk = 0; kk < 64; kk += 32) {
      bf16x16 aq = frag_a(Qs, wave * 16, kk, 64);
      s[0] = wmma_bf16(aq, frag_b(Ks, 0,  kk, 64), s[0]);
      s[1] = wmma_bf16(aq, frag_b(Ks, 16, kk, 64), s[1]);
    }

    // scale + causal mask (only the diagonal tile actually masks)
    const int qrow_base = q0 + wave * 16 + hh * 8;
#pragma unroll
    for (int j = 0; j < 2; ++j)
#pragma unroll
      for (int r = 0; r < 8; ++r) {
        float sv = s[j][r] * scale;
        int kpos = k0 + j * 16 + nn;
        if (kpos > qrow_base + r) sv = -3.0e38f;
        s[j][r] = sv;
      }

    // online softmax (row = hh*8+r; reduce across the 16-lane group)
    float alpha[8];
#pragma unroll
    for (int r = 0; r < 8; ++r) {
      float mx = fmaxf(s[0][r], s[1][r]);
#pragma unroll
      for (int off = 1; off < 16; off <<= 1) mx = fmaxf(mx, __shfl_xor(mx, off, 32));
      float mnew = fmaxf(m[r], mx);
      alpha[r] = __expf(m[r] - mnew);
      m[r] = mnew;
      float p0 = __expf(s[0][r] - mnew);
      float p1 = __expf(s[1][r] - mnew);
      s[0][r] = p0; s[1][r] = p1;
      float rs = p0 + p1;
#pragma unroll
      for (int off = 1; off < 16; off <<= 1) rs += __shfl_xor(rs, off, 32);
      l[r] = l[r] * alpha[r] + rs;
    }

    // P (f32 C-layout) -> bf16 row-major [16][32] in per-wave LDS scratch
    unsigned short* Pw = &Ps[wave][0];
#pragma unroll
    for (int j = 0; j < 2; ++j)
#pragma unroll
      for (int r = 0; r < 8; ++r)
        Pw[(hh * 8 + r) * 32 + j * 16 + nn] = f2bf(s[j][r]);

#pragma unroll
    for (int j = 0; j < 4; ++j)
#pragma unroll
      for (int r = 0; r < 8; ++r) o[j][r] *= alpha[r];

    // O += P * V   (P: 16x32, V: 32x64 row-major [key][d])
    bf16x16 pa = frag_a(Pw, 0, 0, 32);
#pragma unroll
    for (int j = 0; j < 4; ++j)
      o[j] = wmma_bf16(pa, frag_b_kn(Vs, j * 16, 64), o[j]);
  }

  // epilogue: normalize and store bf16 in (B, T, H*D)
  unsigned short* op = O + ((size_t)b * T_ + q0 + wave * 16 + hh * 8) * HID_ + h * D_ + nn;
#pragma unroll
  for (int j = 0; j < 4; ++j)
#pragma unroll
    for (int r = 0; r < 8; ++r)
      op[(size_t)r * HID_ + j * 16] = f2bf(o[j][r] / l[r]);
}

// ---------------------------------------------------------------------------
extern "C" void kernel_launch(void* const* d_in, const int* in_sizes, int n_in,
                              void* d_out, int out_size, void* d_ws, size_t ws_size,
                              hipStream_t stream) {
  (void)in_sizes; (void)n_in; (void)out_size; (void)ws_size;
  const float* X    = (const float*)d_in[0];
  const float* cosp = (const float*)d_in[1];
  const float* sinp = (const float*)d_in[2];
  const float* Wq   = (const float*)d_in[3];
  const float* Wk   = (const float*)d_in[4];
  const float* Wv   = (const float*)d_in[5];
  const float* Wo   = (const float*)d_in[6];
  const float* qw   = (const float*)d_in[7];
  const float* kw   = (const float*)d_in[8];
  float* out = (float*)d_out;

  const size_t MB = 1ull << 20;
  const size_t MT = (size_t)B_ * T_;            // 4096 token rows
  char* ws = (char*)d_ws;
  unsigned short* Xb  = (unsigned short*)(ws);              //  8 MB  MT*1024 bf16
  unsigned short* Wqb = (unsigned short*)(ws + 8  * MB);    //  2 MB
  unsigned short* Wkb = (unsigned short*)(ws + 10 * MB);    //  0.5 MB
  unsigned short* Wvb = (unsigned short*)(ws + 10 * MB + 512 * 1024);
  unsigned short* Wob = (unsigned short*)(ws + 11 * MB);    //  2 MB
  float* qf = (float*)(ws + 13 * MB);                       // 16 MB  MT*1024 f32
  float* kf = (float*)(ws + 29 * MB);                       //  4 MB
  float* vf = (float*)(ws + 33 * MB);                       //  4 MB
  unsigned short* Qb  = (unsigned short*)(ws + 37 * MB);    //  8 MB
  unsigned short* Kb  = (unsigned short*)(ws + 45 * MB);    //  2 MB
  unsigned short* Vb  = (unsigned short*)(ws + 47 * MB);    //  2 MB
  unsigned short* Ofb = (unsigned short*)qf;                // reuse (8 MB of 16)

  dim3 blk(256);
  // one-time f32 -> bf16 conversions
  cvt_f32_bf16<<<dim3((int)(MT * HID_ / 4 / 256)), blk, 0, stream>>>(X,  Xb,  (int)(MT * HID_ / 4));
  cvt_f32_bf16<<<dim3(HID_ * HID_ / 4 / 256), blk, 0, stream>>>(Wq, Wqb, HID_ * HID_ / 4);
  cvt_f32_bf16<<<dim3(HK_ * D_ * HID_ / 4 / 256), blk, 0, stream>>>(Wk, Wkb, HK_ * D_ * HID_ / 4);
  cvt_f32_bf16<<<dim3(HK_ * D_ * HID_ / 4 / 256), blk, 0, stream>>>(Wv, Wvb, HK_ * D_ * HID_ / 4);
  cvt_f32_bf16<<<dim3(HID_ * HID_ / 4 / 256), blk, 0, stream>>>(Wo, Wob, HID_ * HID_ / 4);

  // QKV projections (bf16 WMMA, f32 out)
  gemm_bf16_wmma<<<dim3(MT / 128, HID_ / 64), blk, 0, stream>>>(Xb, Wqb, qf, (int)MT, HID_, HID_);
  gemm_bf16_wmma<<<dim3(MT / 128, (HK_ * D_) / 64), blk, 0, stream>>>(Xb, Wkb, kf, (int)MT, HK_ * D_, HID_);
  gemm_bf16_wmma<<<dim3(MT / 128, (HK_ * D_) / 64), blk, 0, stream>>>(Xb, Wvb, vf, (int)MT, HK_ * D_, HID_);

  int totalRows = B_ * T_ * (H_ + 2 * HK_);     // 98304
  norm_rope_relayout<<<dim3((totalRows + 255) / 256), blk, 0, stream>>>(
      qf, kf, vf, cosp, sinp, qw, kw, Qb, Kb, Vb);

  attn_wmma<<<dim3(T_ / 128, B_ * H_), blk, 0, stream>>>(Qb, Kb, Vb, Ofb);

  gemm_bf16_wmma<<<dim3(MT / 128, HID_ / 64), blk, 0, stream>>>(Ofb, Wob, out, (int)MT, HID_, HID_);
}
